// K_Genreformer_layer_18013092839769
// MI455X (gfx1250) — matrix operand order repaired and checked by reading, not compiled, not run on hardware
//
#include <hip/hip_runtime.h>
#include <hip/hip_bf16.h>
#include <math.h>

typedef __attribute__((ext_vector_type(16))) _Float16 v16h;
typedef __attribute__((ext_vector_type(8)))  float    v8f;
typedef __attribute__((ext_vector_type(2)))  __fp16   f16x2;

static constexpr int DM     = 128;   // model dim
static constexpr int GH     = 4;     // GAT heads
static constexpr int CHD    = 32;    // channels per head
static constexpr int NG     = 30;    // genres
static constexpr int SEQ    = 680;   // L_MAX
static constexpr int TH     = 4;     // transformer heads
static constexpr int FF     = 2048;
static constexpr int NL     = 4;
static constexpr int NHOPS  = 3;
static constexpr int FCHUNK = 4080;  // FFN row chunk (multiple of 16)

// ---------------------------------------------------------------------------
// WMMA GEMM:  C[M,N] = beta*Cin + alpha * act( A[M,K] @ B[N,K]^T + bias[N] )
// act: 0 none, 1 leaky(0.01), 2 relu
// Block = 128 threads = 4 waves; block tile 64x64, K step 32.
// Fragment layouts follow CDNA5 ISA 7.12.2 (wave32).
// ---------------------------------------------------------------------------
#define TM 64
#define TN 64
#define TKK 32

__launch_bounds__(128)
__global__ void k_gemm(const float* __restrict__ A, const float* __restrict__ B,
                       const float* __restrict__ bias, const float* __restrict__ Cin,
                       float* __restrict__ C, int M, int N, int K,
                       float alpha, float beta, int act)
{
  __shared__ _Float16 As[TM][TKK];
  __shared__ _Float16 Bs[TN][TKK];
  const int tid  = threadIdx.x;
  const int wave = tid >> 5;
  const int lane = tid & 31;
  const int m0 = blockIdx.x * TM;
  const int n0 = blockIdx.y * TN;

  v8f acc[4];
#pragma unroll
  for (int t = 0; t < 4; t++)
#pragma unroll
    for (int j = 0; j < 8; j++) acc[t][j] = 0.f;

  for (int k0 = 0; k0 < K; k0 += TKK) {
    const bool fullK = (k0 + TKK <= K);
    // ---- stage A tile ----
    if ((m0 + TM <= M) && fullK) {
      for (int i = tid; i < (TM * TKK) / 4; i += 128) {
        int r = i >> 3, c4 = (i & 7) * 4;
        const float* ap = A + (size_t)(m0 + r) * K + k0 + c4;
        float4 v = *(const float4*)ap;
        *(f16x2*)&As[r][c4]     = __builtin_amdgcn_cvt_pkrtz(v.x, v.y);
        *(f16x2*)&As[r][c4 + 2] = __builtin_amdgcn_cvt_pkrtz(v.z, v.w);
        if (k0 + TKK < K) __builtin_prefetch(ap + TKK, 0, 1);  // global_prefetch_b8
      }
    } else {
      for (int i = tid; i < TM * TKK; i += 128) {
        int r = i >> 5, c = i & 31;
        int gm = m0 + r, gk = k0 + c;
        As[r][c] = (_Float16)((gm < M && gk < K) ? A[(size_t)gm * K + gk] : 0.f);
      }
    }
    // ---- stage B tile ----
    if ((n0 + TN <= N) && fullK) {
      for (int i = tid; i < (TN * TKK) / 4; i += 128) {
        int r = i >> 3, c4 = (i & 7) * 4;
        const float* bp = B + (size_t)(n0 + r) * K + k0 + c4;
        float4 v = *(const float4*)bp;
        *(f16x2*)&Bs[r][c4]     = __builtin_amdgcn_cvt_pkrtz(v.x, v.y);
        *(f16x2*)&Bs[r][c4 + 2] = __builtin_amdgcn_cvt_pkrtz(v.z, v.w);
        if (k0 + TKK < K) __builtin_prefetch(bp + TKK, 0, 1);
      }
    } else {
      for (int i = tid; i < TN * TKK; i += 128) {
        int r = i >> 5, c = i & 31;
        int gn = n0 + r, gk = k0 + c;
        Bs[r][c] = (_Float16)((gn < N && gk < K) ? B[(size_t)gn * K + gk] : 0.f);
      }
    }
    __syncthreads();

    const int mrow  = wave * 16 + (lane & 15);
    const int khalf = (lane >> 4) * 8;
    v16h af;
#pragma unroll
    for (int e = 0; e < 8; e++) af[e]     = As[mrow][khalf + e];
#pragma unroll
    for (int e = 0; e < 8; e++) af[8 + e] = As[mrow][16 + khalf + e];

    const int kb = (lane >> 4) * 8;
    const int nc = lane & 15;
#pragma unroll
    for (int nt = 0; nt < 4; nt++) {
      v16h bf;
#pragma unroll
      for (int e = 0; e < 16; e++) bf[e] = Bs[nt * 16 + nc][(e < 8) ? (kb + e) : (16 + kb + e - 8)];
      acc[nt] = __builtin_amdgcn_wmma_f32_16x16x32_f16(
          false, af, false, bf, (short)0, acc[nt], false, false);
    }
    __syncthreads();
  }

  const int mBase = m0 + wave * 16 + (lane >> 4) * 8;
  const int nj    = lane & 15;
#pragma unroll
  for (int nt = 0; nt < 4; nt++) {
    int gn = n0 + nt * 16 + nj;
    if (gn >= N) continue;
    float bv = bias ? bias[gn] : 0.f;
#pragma unroll
    for (int j = 0; j < 8; j++) {
      int gm = mBase + j;
      if (gm >= M) continue;
      float v = acc[nt][j] + bv;
      if (act == 1)      v = (v >= 0.f) ? v : 0.01f * v;
      else if (act == 2) v = fmaxf(v, 0.f);
      float o = alpha * v;
      if (Cin) o += beta * Cin[(size_t)gm * N + gn];
      C[(size_t)gm * N + gn] = o;
    }
  }
}

// ---------------------------------------------------------------------------
// Flash-attention style WMMA kernel.  One wave per (genre, head, 16-query tile).
// scores = (Q/sqrt(32)) @ K^T via wmma (head dim 32 == one WMMA K step),
// online softmax with shfl_xor row reductions, P tile staged through LDS to
// reshape C-layout -> A-fragment layout, then o += P @ V via wmma.
// ---------------------------------------------------------------------------
__launch_bounds__(32)
__global__ void k_attn_wmma(const float* __restrict__ qkv, const int* __restrict__ counts,
                            float* __restrict__ ao)
{
  constexpr int QT = (SEQ + 15) / 16;   // 43 query tiles
  constexpr int KT = (SEQ + 31) / 32;   // 22 key tiles (32 keys each)

  const int b    = blockIdx.x;
  const int qt   = b % QT;
  const int gh   = b / QT;
  const int h    = gh % TH;
  const int g    = gh / TH;
  const int lane = threadIdx.x;
  const int half = lane >> 4;
  const int l16  = lane & 15;

  __shared__ _Float16 Kt[32][32];
  __shared__ _Float16 Vt[32][32];
  __shared__ _Float16 Pt[16][32];

  const long base = (long)g * SEQ;
  const int  cnt  = counts[g];
  const int  q0   = qt * 16;

  // Q fragment (A layout), 1/sqrt(32) folded in
  v16h qf;
  {
    const int row = q0 + l16;
    const bool ok = row < SEQ;
    const float* qp = qkv + (base + (ok ? row : 0)) * (3 * DM) + h * 32;
    const int kh = half * 8;
#pragma unroll
    for (int e = 0; e < 8; e++) {
      float v0 = ok ? qp[kh + e]      : 0.f;
      float v1 = ok ? qp[16 + kh + e] : 0.f;
      qf[e]     = (_Float16)(v0 * 0.17677669529663687f);
      qf[8 + e] = (_Float16)(v1 * 0.17677669529663687f);
    }
  }

  float m[8], l[8];
  v8f o0, o1;
#pragma unroll
  for (int j = 0; j < 8; j++) { m[j] = -3.0e38f; l[j] = 0.f; o0[j] = 0.f; o1[j] = 0.f; }

  for (int kt = 0; kt < KT; kt++) {
    const int kbase = kt * 32;
    // cooperative K/V tile load: lane -> one key row (32 channels)
    {
      const int key = kbase + lane;
      const bool ok = key < cnt;    // padded / masked keys load as zero
      const float* kp = qkv + (base + (ok ? key : 0)) * (3 * DM) + DM + h * 32;
      const float* vp = kp + DM;
#pragma unroll
      for (int c = 0; c < 32; c += 4) {
        float4 kv, vv;
        if (ok) { kv = *(const float4*)(kp + c); vv = *(const float4*)(vp + c); }
        else    { kv = make_float4(0.f,0.f,0.f,0.f); vv = kv; }
        *(f16x2*)&Kt[lane][c]     = __builtin_amdgcn_cvt_pkrtz(kv.x, kv.y);
        *(f16x2*)&Kt[lane][c + 2] = __builtin_amdgcn_cvt_pkrtz(kv.z, kv.w);
        *(f16x2*)&Vt[lane][c]     = __builtin_amdgcn_cvt_pkrtz(vv.x, vv.y);
        *(f16x2*)&Vt[lane][c + 2] = __builtin_amdgcn_cvt_pkrtz(vv.z, vv.w);
      }
    }
    __syncthreads();

    // scores: two 16-key sub-tiles  (B layout: lane -> key=l16, ch e<8 ? half*8+e : 16+half*8+e-8)
    v8f s0, s1, z;
#pragma unroll
    for (int j = 0; j < 8; j++) z[j] = 0.f;
    {
      v16h bf;
#pragma unroll
      for (int e = 0; e < 16; e++) bf[e] = Kt[l16][(e < 8) ? (half * 8 + e) : (16 + half * 8 + e - 8)];
      s0 = __builtin_amdgcn_wmma_f32_16x16x32_f16(false, qf, false, bf, (short)0, z, false, false);
#pragma unroll
      for (int e = 0; e < 16; e++) bf[e] = Kt[16 + l16][(e < 8) ? (half * 8 + e) : (16 + half * 8 + e - 8)];
      s1 = __builtin_amdgcn_wmma_f32_16x16x32_f16(false, qf, false, bf, (short)0, z, false, false);
    }

    const bool ok0 = (kbase + l16) < cnt;
    const bool ok1 = (kbase + 16 + l16) < cnt;

    // online softmax update; C layout: lane holds key=l16, rows 8*half+j
#pragma unroll
    for (int j = 0; j < 8; j++) {
      float a0 = ok0 ? s0[j] : -1e9f;
      float a1 = ok1 ? s1[j] : -1e9f;
      float t = fmaxf(a0, a1);
      t = fmaxf(t, __shfl_xor(t, 1, 32));
      t = fmaxf(t, __shfl_xor(t, 2, 32));
      t = fmaxf(t, __shfl_xor(t, 4, 32));
      t = fmaxf(t, __shfl_xor(t, 8, 32));
      float mn = fmaxf(m[j], t);
      float sc = __expf(m[j] - mn);
      float p0 = __expf(a0 - mn);
      float p1 = __expf(a1 - mn);
      float ps = p0 + p1;
      ps += __shfl_xor(ps, 1, 32);
      ps += __shfl_xor(ps, 2, 32);
      ps += __shfl_xor(ps, 4, 32);
      ps += __shfl_xor(ps, 8, 32);
      l[j] = l[j] * sc + ps;
      m[j] = mn;
      o0[j] *= sc;
      o1[j] *= sc;
      Pt[8 * half + j][l16]      = (_Float16)p0;
      Pt[8 * half + j][16 + l16] = (_Float16)p1;
    }
    __syncthreads();

    // o += P @ V : A fragment from Pt, B fragments from Vt
    v16h pf;
    {
      const int kh = half * 8;
#pragma unroll
      for (int e = 0; e < 8; e++) {
        pf[e]     = Pt[l16][kh + e];
        pf[8 + e] = Pt[l16][16 + kh + e];
      }
    }
    {
      v16h vf;
#pragma unroll
      for (int e = 0; e < 16; e++) vf[e] = Vt[(e < 8) ? (half * 8 + e) : (16 + half * 8 + e - 8)][l16];
      o0 = __builtin_amdgcn_wmma_f32_16x16x32_f16(false, pf, false, vf, (short)0, o0, false, false);
#pragma unroll
      for (int e = 0; e < 16; e++) vf[e] = Vt[(e < 8) ? (half * 8 + e) : (16 + half * 8 + e - 8)][16 + l16];
      o1 = __builtin_amdgcn_wmma_f32_16x16x32_f16(false, pf, false, vf, (short)0, o1, false, false);
    }
    __syncthreads();
  }

  // write out: lane holds ch=l16 (o0) / 16+l16 (o1), rows q0 + 8*half + j
#pragma unroll
  for (int j = 0; j < 8; j++) {
    int row = q0 + 8 * half + j;
    if (row < SEQ) {
      float inv = 1.f / l[j];
      long orow = (base + row) * DM + h * 32;
      ao[orow + l16]      = o0[j] * inv;
      ao[orow + 16 + l16] = o1[j] * inv;
    }
  }
}

// ---------------------------------------------------------------------------
// small utility kernels
// ---------------------------------------------------------------------------
__global__ void k_fill(float* p, float v, long n) {
  long i  = (long)blockIdx.x * blockDim.x + threadIdx.x;
  long st = (long)gridDim.x * blockDim.x;
  for (; i < n; i += st) p[i] = v;
}

__global__ void k_copy(float* dst, const float* src, long n) {
  long i = (long)blockIdx.x * blockDim.x + threadIdx.x;
  if (i < n) dst[i] = src[i];
}

__global__ void k_bias_rows(float* p, const float* __restrict__ bias, long n) {
  long i = (long)blockIdx.x * blockDim.x + threadIdx.x;
  if (i < n) p[i] = bias[i & (DM - 1)];
}

__global__ void k_node_logits(const float* __restrict__ h,
                              const float* __restrict__ asrc,
                              const float* __restrict__ adst,
                              float* __restrict__ als, float* __restrict__ ald, int N)
{
  long t = (long)blockIdx.x * blockDim.x + threadIdx.x;
  if (t >= (long)N * GH) return;
  int n = (int)(t >> 2), hd = (int)(t & 3);
  const float* hp = h + (size_t)n * DM + hd * CHD;
  float s = 0.f, d = 0.f;
  for (int c = 0; c < CHD; c++) { s += hp[c] * asrc[hd * CHD + c]; d += hp[c] * adst[hd * CHD + c]; }
  als[t] = s; ald[t] = d;
}

__device__ inline float atomicMaxF(float* addr, float val) {
  if (val >= 0.f)
    return __int_as_float(atomicMax((int*)addr, __float_as_int(val)));
  else
    return __uint_as_float(atomicMin((unsigned int*)addr, __float_as_uint(val)));
}

__device__ inline void edge_sd(const int* __restrict__ ei, int E, long e, int& s, int& d) {
  if (e < E) { s = ei[e]; d = ei[(long)E + e]; } else { s = d = (int)(e - E); }
}

__global__ void k_edge_max(const int* __restrict__ ei, int E, int N,
                           const float* __restrict__ als, const float* __restrict__ ald,
                           float* __restrict__ mmax)
{
  long t = (long)blockIdx.x * blockDim.x + threadIdx.x;
  if (t >= (long)(E + N) * GH) return;
  long e = t >> 2; int hd = (int)(t & 3);
  int s, d; edge_sd(ei, E, e, s, d);
  float v = als[(long)s * GH + hd] + ald[(long)d * GH + hd];
  v = (v >= 0.f) ? v : 0.2f * v;
  atomicMaxF(&mmax[(long)d * GH + hd], v);
}

__global__ void k_edge_exp(const int* __restrict__ ei, int E, int N,
                           const float* __restrict__ als, const float* __restrict__ ald,
                           const float* __restrict__ mmax,
                           float* __restrict__ ee, float* __restrict__ den)
{
  long t = (long)blockIdx.x * blockDim.x + threadIdx.x;
  if (t >= (long)(E + N) * GH) return;
  long e = t >> 2; int hd = (int)(t & 3);
  int s, d; edge_sd(ei, E, e, s, d);
  float v = als[(long)s * GH + hd] + ald[(long)d * GH + hd];
  v = (v >= 0.f) ? v : 0.2f * v;
  float x = __expf(v - mmax[(long)d * GH + hd]);
  ee[t] = x;
  atomicAdd(&den[(long)d * GH + hd], x);
}

__global__ void k_edge_scatter(const int* __restrict__ ei, int E, int N,
                               const float* __restrict__ hfeat,
                               const float* __restrict__ ee, const float* __restrict__ den,
                               float* __restrict__ gout)
{
  long t = (long)blockIdx.x * blockDim.x + threadIdx.x;
  if (t >= (long)(E + N) * DM) return;
  long e = t >> 7; int c = (int)(t & 127); int hd = c >> 5;
  int s, d; edge_sd(ei, E, e, s, d);
  float alpha = ee[e * GH + hd] / den[(long)d * GH + hd];
  atomicAdd(&gout[(size_t)d * DM + c], hfeat[(size_t)s * DM + c] * alpha);
}

__global__ void k_pos(const int* __restrict__ genre, int* __restrict__ pos,
                      int* __restrict__ counts, int NM, int ngen)
{
  int g = threadIdx.x;
  if (g >= ngen) return;
  int c = 0;
  for (int i = 0; i < NM; i++)
    if (genre[i] == g) pos[i] = c++;
  counts[g] = c;
}

__global__ void k_scatter_seq(const float* __restrict__ src, const int* __restrict__ midx,
                              const int* __restrict__ genre, const int* __restrict__ pos,
                              float* __restrict__ y, int NM)
{
  long t = (long)blockIdx.x * blockDim.x + threadIdx.x;
  if (t >= (long)NM * DM) return;
  int m = (int)(t / DM), c = (int)(t % DM);
  long row = (long)genre[m] * SEQ + pos[m];
  y[row * DM + c] = src[(size_t)midx[m] * DM + c];
}

__launch_bounds__(128)
__global__ void k_add_ln(const float* __restrict__ x, const float* __restrict__ r,
                         const float* __restrict__ gam, const float* __restrict__ bet,
                         float* __restrict__ y)
{
  long row = blockIdx.x;
  int tid = threadIdx.x;
  __shared__ float red[128];
  float v = x[row * DM + tid] + r[row * DM + tid];
  red[tid] = v; __syncthreads();
  for (int st = 64; st > 0; st >>= 1) { if (tid < st) red[tid] += red[tid + st]; __syncthreads(); }
  float mean = red[0] * (1.f / DM);
  __syncthreads();
  float dv = v - mean;
  red[tid] = dv * dv; __syncthreads();
  for (int st = 64; st > 0; st >>= 1) { if (tid < st) red[tid] += red[tid + st]; __syncthreads(); }
  float var = red[0] * (1.f / DM);
  y[row * DM + tid] = dv * rsqrtf(var + 1e-5f) * gam[tid] + bet[tid];
}

__global__ void k_pool(const float* __restrict__ y, const int* __restrict__ counts,
                       float* __restrict__ pooled)
{
  long t = (long)blockIdx.x * blockDim.x + threadIdx.x;
  if (t >= (long)NG * DM) return;
  int g = (int)(t / DM), c = (int)(t % DM);
  int cnt = counts[g];
  float s = 0.f;
  for (int p = 0; p < cnt; p++) s += y[((long)g * SEQ + p) * DM + c];
  pooled[t] = s / fmaxf((float)cnt, 1.f);
}

__global__ void k_build_cat(const float* __restrict__ src, const int* __restrict__ midx,
                            const int* __restrict__ genre, const float* __restrict__ pooled,
                            float* __restrict__ cat, int NM)
{
  long t = (long)blockIdx.x * blockDim.x + threadIdx.x;
  if (t >= (long)NM * 2 * DM) return;
  int m = (int)(t / (2 * DM)), c = (int)(t % (2 * DM));
  cat[t] = (c < DM) ? src[(size_t)midx[m] * DM + c]
                    : pooled[(long)genre[m] * DM + (c - DM)];
}

__global__ void k_scatter_out(float* __restrict__ out, const float* __restrict__ fus,
                              const int* __restrict__ midx, int NM)
{
  long t = (long)blockIdx.x * blockDim.x + threadIdx.x;
  if (t >= (long)NM * DM) return;
  int m = (int)(t / DM), c = (int)(t % DM);
  out[(size_t)midx[m] * DM + c] = fus[t];
}

// ---------------------------------------------------------------------------
extern "C" void kernel_launch(void* const* d_in, const int* in_sizes, int n_in,
                              void* d_out, int out_size, void* d_ws, size_t ws_size,
                              hipStream_t stream)
{
  const float* x     = (const float*)d_in[0];
  const int*   eidx[3] = {(const int*)d_in[1], (const int*)d_in[2], (const int*)d_in[3]};
  const int*   midx  = (const int*)d_in[4];
  const int*   genre = (const int*)d_in[5];
  const float* gatW  = (const float*)d_in[6];
  const float* attS  = (const float*)d_in[7];
  const float* attD  = (const float*)d_in[8];
  const float* gatB  = (const float*)d_in[9];
  const float* decW  = (const float*)d_in[10];
  const float* decB  = (const float*)d_in[11];
  const float* qkvW  = (const float*)d_in[12];
  const float* qkvB  = (const float*)d_in[13];
  const float* outW  = (const float*)d_in[14];
  const float* outB  = (const float*)d_in[15];
  const float* ln1g  = (const float*)d_in[16];
  const float* ln1b  = (const float*)d_in[17];
  const float* ln2g  = (const float*)d_in[18];
  const float* ln2b  = (const float*)d_in[19];
  const float* ffw1  = (const float*)d_in[20];
  const float* ffb1  = (const float*)d_in[21];
  const float* ffw2  = (const float*)d_in[22];
  const float* ffb2  = (const float*)d_in[23];
  const float* fusW  = (const float*)d_in[24];
  const float* fusB  = (const float*)d_in[25];
  (void)n_in; (void)out_size; (void)ws_size;

  const int NN = in_sizes[0] / DM;   // 50000 nodes
  const int E  = in_sizes[1] / 2;    // 400000 edges
  const int NM = in_sizes[4];        // 20400 movies
  const int NT = NG * SEQ;           // 20400 tokens

  char* wbase = (char*)d_ws;
  size_t off = 0;
  auto walloc = [&](size_t bytes) -> char* {
    char* p = wbase + off;
    off += (bytes + 255) & ~(size_t)255;
    return p;
  };
  float* hbuf = (float*)walloc((size_t)NN * DM * 4);
  float* gout = (float*)walloc((size_t)NN * DM * 4);
  float* acc  = (float*)walloc((size_t)NN * DM * 4);
  float* als  = (float*)walloc((size_t)NN * GH * 4);
  float* ald  = (float*)walloc((size_t)NN * GH * 4);
  float* mmax = (float*)walloc((size_t)NN * GH * 4);
  float* den  = (float*)walloc((size_t)NN * GH * 4);
  float* ee   = (float*)walloc((size_t)(E + NN) * GH * 4);
  int*   pos  = (int*)  walloc((size_t)NM * 4);
  int*   cnts = (int*)  walloc((size_t)NG * 4);
  float* y    = (float*)walloc((size_t)NT * DM * 4);
  float* t1   = (float*)walloc((size_t)NT * DM * 4);
  float* qkv  = (float*)walloc((size_t)NT * 3 * DM * 4);
  float* ao   = (float*)walloc((size_t)NT * DM * 4);
  float* mid  = (float*)walloc((size_t)FCHUNK * FF * 4);
  float* pool = (float*)walloc((size_t)NG * DM * 4);
  float* cat  = (float*)walloc((size_t)NM * 2 * DM * 4);
  float* fus  = (float*)walloc((size_t)NM * DM * 4);

  const int TPB = 256;
  auto nblk = [](long n, int t) -> unsigned { return (unsigned)((n + t - 1) / t); };
  auto gemm = [&](const float* A, const float* B, const float* bias, const float* Cin,
                  float* C, int M, int N, int K, float alpha, float beta, int act) {
    dim3 g((M + TM - 1) / TM, (N + TN - 1) / TN);
    k_gemm<<<g, dim3(128), 0, stream>>>(A, B, bias, Cin, C, M, N, K, alpha, beta, act);
  };

  // ---------------- multi-hop GAT ----------------
  for (int k = 0; k < NHOPS; k++) {
    gemm(x, gatW + (size_t)k * DM * DM, nullptr, nullptr, hbuf, NN, DM, DM, 1.f, 0.f, 0);
    k_node_logits<<<nblk((long)NN * GH, TPB), TPB, 0, stream>>>(
        hbuf, attS + (size_t)k * GH * CHD, attD + (size_t)k * GH * CHD, als, ald, NN);
    k_fill<<<1024, TPB, 0, stream>>>(mmax, -3.0e38f, (long)NN * GH);
    k_fill<<<1024, TPB, 0, stream>>>(den, 0.f, (long)NN * GH);
    k_bias_rows<<<nblk((long)NN * DM, TPB), TPB, 0, stream>>>(gout, gatB + (size_t)k * DM, (long)NN * DM);
    long tote = (long)(E + NN) * GH;
    k_edge_max<<<nblk(tote, TPB), TPB, 0, stream>>>(eidx[k], E, NN, als, ald, mmax);
    k_edge_exp<<<nblk(tote, TPB), TPB, 0, stream>>>(eidx[k], E, NN, als, ald, mmax, ee, den);
    long totc = (long)(E + NN) * DM;
    k_edge_scatter<<<nblk(totc, TPB), TPB, 0, stream>>>(eidx[k], E, NN, hbuf, ee, den, gout);
    float a = expf(-0.1f * (float)k);
    gemm(gout, decW + (size_t)k * DM * DM, decB + (size_t)k * DM,
         (k == 0 ? nullptr : acc), acc, NN, DM, DM, a, 1.f, 1);
  }

  // ---------------- pack per-genre sequences ----------------
  k_pos<<<1, 32, 0, stream>>>(genre, pos, cnts, NM, NG);
  k_fill<<<1024, TPB, 0, stream>>>(y, 0.f, (long)NT * DM);
  k_scatter_seq<<<nblk((long)NM * DM, TPB), TPB, 0, stream>>>(acc, midx, genre, pos, y, NM);

  // ---------------- transformer encoder ----------------
  constexpr int QT = (SEQ + 15) / 16;
  for (int l = 0; l < NL; l++) {
    gemm(y, qkvW + (size_t)l * 3 * DM * DM, qkvB + (size_t)l * 3 * DM, nullptr,
         qkv, NT, 3 * DM, DM, 1.f, 0.f, 0);
    k_attn_wmma<<<NG * TH * QT, 32, 0, stream>>>(qkv, cnts, ao);
    gemm(ao, outW + (size_t)l * DM * DM, outB + (size_t)l * DM, nullptr,
         t1, NT, DM, DM, 1.f, 0.f, 0);
    k_add_ln<<<NT, DM, 0, stream>>>(y, t1, ln1g + (size_t)l * DM, ln1b + (size_t)l * DM, y);
    for (int r0 = 0; r0 < NT; r0 += FCHUNK) {
      int mr = (NT - r0 < FCHUNK) ? (NT - r0) : FCHUNK;
      gemm(y + (size_t)r0 * DM, ffw1 + (size_t)l * FF * DM, ffb1 + (size_t)l * FF, nullptr,
           mid, mr, FF, DM, 1.f, 0.f, 2);
      gemm(mid, ffw2 + (size_t)l * DM * FF, ffb2 + (size_t)l * DM, nullptr,
           t1 + (size_t)r0 * DM, mr, DM, FF, 1.f, 0.f, 0);
    }
    k_add_ln<<<NT, DM, 0, stream>>>(y, t1, ln2g + (size_t)l * DM, ln2b + (size_t)l * DM, y);
  }

  // ---------------- pool, fuse, write output ----------------
  k_pool<<<nblk((long)NG * DM, TPB), TPB, 0, stream>>>(y, cnts, pool);
  k_build_cat<<<nblk((long)NM * 2 * DM, TPB), TPB, 0, stream>>>(acc, midx, genre, pool, cat, NM);
  gemm(cat, fusW, fusB, nullptr, fus, NM, DM, 2 * DM, 1.f, 0.f, 1);

  float* out = (float*)d_out;
  k_copy<<<nblk((long)NN * DM, TPB), TPB, 0, stream>>>(out, acc, (long)NN * DM);
  k_scatter_out<<<nblk((long)NM * DM, TPB), TPB, 0, stream>>>(out, fus, midx, NM);
}
